// PNA_61976378081729
// MI455X (gfx1250) — matrix operand order repaired and checked by reading, not compiled, not run on hardware
//
#include <hip/hip_runtime.h>
#include <hip/hip_bf16.h>

typedef _Float16 v16h __attribute__((ext_vector_type(16)));
typedef _Float16 v8h  __attribute__((ext_vector_type(8)));
typedef float    v8f  __attribute__((ext_vector_type(8)));

#define DEG 10
#define F1 20
#define NODES_PER_BLK 16
#define EDGES_PER_BLK (NODES_PER_BLK * DEG)   // 160
#define KE 64    // edge GEMM K: 40 feats + bias row @40, padded to 64
#define KP 128   // tower GEMM K: 100 feats + bias row @100, padded to 128
#define MPAD 32  // padded N stride for LDS C buffers

// workspace layout (byte offsets)
#define WS_W1C   0u         // 300 f32  (folded layer-1 tower, [20 x 15])
#define WS_B1C   2048u      // 20 f32
#define WS_BPREF 4096u      // 2048 f16: w_pre2^T WMMA B-frags (2 kchunks x 2 ntiles x 32 lanes x 16)
#define WS_W2CF  8192u      // 4096 f16: folded tower WMMA B-frags (4 kchunks x 2 ntiles x 32 lanes x 16)
#define WS_H1    32768u     // N*20 f16

// ---------------------------------------------------------------------------
// Prep: fold degree-scalers (all == 1 for constant deg=10) + w_lin into the
// post weights; fold biases in as an extra K-row of B; pack WMMA B fragments.
// ---------------------------------------------------------------------------
__global__ void pna_prep(const float* __restrict__ w_post1, const float* __restrict__ b_post1,
                         const float* __restrict__ w_lin1,  const float* __restrict__ b_lin1,
                         const float* __restrict__ w_pre2,  const float* __restrict__ b_pre2,
                         const float* __restrict__ w_post2, const float* __restrict__ b_post2,
                         const float* __restrict__ w_lin2,  const float* __restrict__ b_lin2,
                         float* __restrict__ w1c, float* __restrict__ b1c,
                         _Float16* __restrict__ bpref, _Float16* __restrict__ w2cf)
{
    __shared__ float sW2c[20 * 100];
    __shared__ float sB2c[20];
    const int tid = threadIdx.x;

    // layer-1: w1c = w_lin1 @ fold(w_post1)  -> [20 x 15]
    for (int i = tid; i < 20 * 15; i += blockDim.x) {
        int o = i / 15, k = i % 15;
        float acc = 0.f;
        for (int p = 0; p < 20; ++p) {
            float fp = (k < 3) ? w_post1[p * 51 + k]
                               : (w_post1[p * 51 + k]      + w_post1[p * 51 + k + 12] +
                                  w_post1[p * 51 + k + 24] + w_post1[p * 51 + k + 36]);
            acc += w_lin1[o * 20 + p] * fp;
        }
        w1c[i] = acc;
    }
    // fused biases
    for (int o = tid; o < 20; o += blockDim.x) {
        float a1 = b_lin1[o], a2 = b_lin2[o];
        for (int p = 0; p < 20; ++p) {
            a1 += w_lin1[o * 20 + p] * b_post1[p];
            a2 += w_lin2[o * 20 + p] * b_post2[p];
        }
        b1c[o] = a1; sB2c[o] = a2;
    }
    // layer-2: w2c = w_lin2 @ fold(w_post2) -> [20 x 100]
    for (int i = tid; i < 20 * 100; i += blockDim.x) {
        int o = i / 100, k = i % 100;
        float acc = 0.f;
        for (int p = 0; p < 20; ++p) {
            float fp = (k < 20) ? w_post2[p * 340 + k]
                                : (w_post2[p * 340 + k]       + w_post2[p * 340 + k + 80] +
                                   w_post2[p * 340 + k + 160] + w_post2[p * 340 + k + 240]);
            acc += w_lin2[o * 20 + p] * fp;
        }
        sW2c[i] = acc;
    }
    __syncthreads();

    // Pack B fragments. B order used here (contiguous k, KxN=32x16 per wmma): lanes 0-15 hold
    // K=kchunk*32+(0..15), lanes 16-31 hold K=kchunk*32+(16..31); N = ntile*16 + lane%16.
    // Bias folded in at K==40 (edge) / K==100 (tower); A supplies 1.0 there.
    for (int i = tid; i < 2 * 2 * 32 * 16; i += blockDim.x) {   // w_pre2^T + bias row
        int j = i & 15, lane = (i >> 4) & 31, ct = i >> 9;
        int c = ct >> 1, t = ct & 1;
        int k = 32 * c + ((lane < 16) ? 0 : 16) + j;
        int n = 16 * t + (lane & 15);
        float v = 0.f;
        if (n < 20) {
            if (k < 40)       v = w_pre2[n * 40 + k];
            else if (k == 40) v = b_pre2[n];
        }
        bpref[i] = (_Float16)v;
    }
    for (int i = tid; i < 4 * 2 * 32 * 16; i += blockDim.x) {   // w2c^T + bias row
        int j = i & 15, lane = (i >> 4) & 31, ct = i >> 9;
        int c = ct >> 1, t = ct & 1;
        int k = 32 * c + ((lane < 16) ? 0 : 16) + j;
        int n = 16 * t + (lane & 15);
        float v = 0.f;
        if (n < 20) {
            if (k < 100)       v = sW2c[n * 100 + k];
            else if (k == 100) v = sB2c[n];
        }
        w2cf[i] = (_Float16)v;
    }
}

// ---------------------------------------------------------------------------
// Layer 1: K=6 edge MLP is too small for the matrix pipes -> scalar per node.
// Emits h1 in f16 for the layer-2 WMMA path.
// ---------------------------------------------------------------------------
__global__ void pna_layer1(const float* __restrict__ x, const int* __restrict__ src,
                           const float* __restrict__ w_pre1, const float* __restrict__ b_pre1,
                           const float* __restrict__ w1c, const float* __restrict__ b1c,
                           _Float16* __restrict__ h1, int n_nodes)
{
    int n = blockIdx.x * blockDim.x + threadIdx.x;
    if (n >= n_nodes) return;
    float xi0 = x[n * 3], xi1 = x[n * 3 + 1], xi2 = x[n * 3 + 2];
    float s[3] = {0.f, 0.f, 0.f}, s2[3] = {0.f, 0.f, 0.f};
    float mn[3], mx[3];
    mn[0] = mn[1] = mn[2] = 3.4e38f;
    mx[0] = mx[1] = mx[2] = -3.4e38f;
    #pragma unroll 1
    for (int e = 0; e < DEG; ++e) {
        int sj = src[n * DEG + e];
        float xj0 = x[sj * 3], xj1 = x[sj * 3 + 1], xj2 = x[sj * 3 + 2];
        #pragma unroll
        for (int k = 0; k < 3; ++k) {
            const float* w = &w_pre1[k * 6];
            float m = b_pre1[k] + w[0] * xi0 + w[1] * xi1 + w[2] * xi2
                                + w[3] * xj0 + w[4] * xj1 + w[5] * xj2;
            s[k] += m; s2[k] += m * m;
            mn[k] = fminf(mn[k], m); mx[k] = fmaxf(mx[k], m);
        }
    }
    float v[15];
    v[0] = xi0; v[1] = xi1; v[2] = xi2;
    #pragma unroll
    for (int k = 0; k < 3; ++k) {
        float mean = s[k] * (1.f / DEG);
        float var  = fmaxf(s2[k] * (1.f / DEG) - mean * mean, 0.f);
        v[3 + k] = mean; v[6 + k] = mn[k]; v[9 + k] = mx[k];
        v[12 + k] = sqrtf(var + 1e-5f);
    }
    #pragma unroll 1
    for (int o = 0; o < F1; ++o) {
        float acc = b1c[o];
        #pragma unroll
        for (int k = 0; k < 15; ++k) acc += w1c[o * 15 + k] * v[k];
        h1[n * F1 + o] = (_Float16)fmaxf(acc, 0.f);
    }
}

// A-fragment order used here (contiguous k, 16x32): lanes 0-15 hold K c*32 + 0..15,
// lanes 16-31 hold K c*32 + 16..31, row M = lane%16.
__device__ __forceinline__ v16h load_a_frag(const _Float16* abase, int koff) {
    v8h lo = *(const v8h*)(abase + 2 * koff);
    v8h hi = *(const v8h*)(abase + 2 * koff + 8);
    return __builtin_shufflevector(lo, hi, 0,1,2,3,4,5,6,7,8,9,10,11,12,13,14,15);
}

// ---------------------------------------------------------------------------
// Layer 2 + readout, fully fused: one wave handles 16 nodes (160 contiguous
// edges). Messages live only in LDS. WMMA f16 with f32 accumulation; biases
// ride in the GEMM (K rows 40/100), so the edge epilogue is pure ds_store.
// ---------------------------------------------------------------------------
__launch_bounds__(32)
__global__ void pna_layer2(const int* __restrict__ src, const int* __restrict__ dst,
                           const _Float16* __restrict__ h1,
                           const _Float16* __restrict__ bpref, const _Float16* __restrict__ w2cf,
                           const float* __restrict__ w_out, const float* __restrict__ b_out,
                           float* __restrict__ out)
{
    __shared__ __align__(16) _Float16 sA[EDGES_PER_BLK * KE];     // edge rows [h_dst|h_src|1|0]
    __shared__ __align__(16) float    sM2[EDGES_PER_BLK * MPAD];  // messages (cols 20..31 exact 0)
    __shared__ __align__(16) _Float16 sV[NODES_PER_BLK * KP];     // folded tower input
    __shared__ __align__(16) float    sH2[NODES_PER_BLK * MPAD];  // tower output

    const int lane = threadIdx.x;
    const int m16  = lane & 15;
    const int hi   = lane >> 4;
    const int koff = hi * 8;
    const int nodeBase = blockIdx.x * NODES_PER_BLK;
    const int edgeBase = nodeBase * DEG;

    // speculative prefetch of next block's edge-index segment (safe if OOB)
    __builtin_prefetch(src + edgeBase + EDGES_PER_BLK, 0, 1);
    __builtin_prefetch(dst + edgeBase + EDGES_PER_BLK, 0, 1);

    // ---- stage edge rows into LDS (8B vector copies); K=40 gets the bias 1.0 ----
    for (int r = lane; r < EDGES_PER_BLK; r += 32) {
        int e = edgeBase + r;
        int d = dst[e], sj = src[e];
        const uint2* hd = (const uint2*)(h1 + d  * F1);
        const uint2* hs = (const uint2*)(h1 + sj * F1);
        uint2* row = (uint2*)(sA + r * KE);
        #pragma unroll
        for (int q = 0; q < 5; ++q) row[q] = hd[q];
        #pragma unroll
        for (int q = 0; q < 5; ++q) row[5 + q] = hs[q];
        row[10] = make_uint2(0x3C00u, 0u);                       // K=40 -> 1.0h (bias row)
        #pragma unroll
        for (int q = 11; q < 16; ++q) row[q] = make_uint2(0u, 0u);
    }
    __syncthreads();

    // w_pre2^T fragments: [kchunk][ntile]
    const v16h* bfr = (const v16h*)bpref;
    v16h b00 = bfr[0 * 32 + lane];   // k0, n-tile 0
    v16h b01 = bfr[1 * 32 + lane];   // k0, n-tile 1
    v16h b10 = bfr[2 * 32 + lane];   // k1, n-tile 0 (incl bias row)
    v16h b11 = bfr[3 * 32 + lane];   // k1, n-tile 1 (incl bias row)

    // ---- edge GEMM: 10 M-tiles of 16 edges, K=64, N=32 (20 valid) ----
    #pragma unroll 1
    for (int t = 0; t < EDGES_PER_BLK / 16; ++t) {
        const _Float16* abase = sA + (t * 16 + m16) * KE;
        v16h a0 = load_a_frag(abase, koff);        // K 0..31
        v16h a1 = load_a_frag(abase + 32, koff);   // K 32..63
        v8f c0 = {}; v8f c1 = {};
        c0 = __builtin_amdgcn_wmma_f32_16x16x32_f16(false, a0, false, b00, (short)0, c0, false, false);
        c0 = __builtin_amdgcn_wmma_f32_16x16x32_f16(false, a1, false, b10, (short)0, c0, false, false);
        c1 = __builtin_amdgcn_wmma_f32_16x16x32_f16(false, a0, false, b01, (short)0, c1, false, false);
        c1 = __builtin_amdgcn_wmma_f32_16x16x32_f16(false, a1, false, b11, (short)0, c1, false, false);
        // C layout: VGPR r -> row r (lanes 0-15) / row r+8 (lanes 16-31), col = lane%16 (+16 ntile1)
        // Unconditional scatter: bias is already inside, cols 20..31 are exact zeros.
        #pragma unroll
        for (int r = 0; r < 8; ++r) {
            int row = t * 16 + r + hi * 8;
            sM2[row * MPAD + m16]      = c0[r];
            sM2[row * MPAD + 16 + m16] = c1[r];
        }
    }
    __syncthreads();

    // ---- per-node aggregation over 10 contiguous edges -> folded 100-wide f16 row ----
    for (int idx = lane; idx < NODES_PER_BLK * F1; idx += 32) {
        int i = idx / F1, f = idx % F1;
        const float* mrow = sM2 + (i * DEG) * MPAD + f;
        float sm = 0.f, sq = 0.f, mnv = 3.4e38f, mxv = -3.4e38f;
        #pragma unroll
        for (int e = 0; e < DEG; ++e) {
            float mv = mrow[e * MPAD];
            sm += mv; sq += mv * mv;
            mnv = fminf(mnv, mv); mxv = fmaxf(mxv, mv);
        }
        float mean = sm * (1.f / DEG);
        float var  = fmaxf(sq * (1.f / DEG) - mean * mean, 0.f);
        float stdv = sqrtf(var + 1e-5f);
        _Float16* vrow = sV + i * KP;
        vrow[f]      = h1[(nodeBase + i) * F1 + f];  // self features
        vrow[20 + f] = (_Float16)mean;
        vrow[40 + f] = (_Float16)mnv;
        vrow[60 + f] = (_Float16)mxv;
        vrow[80 + f] = (_Float16)stdv;
    }
    for (int idx = lane; idx < NODES_PER_BLK * 28; idx += 32) {   // K pad: 1.0 at 100, else 0
        int i = idx / 28, f = idx % 28;
        sV[i * KP + 100 + f] = (f == 0) ? (_Float16)1.f : (_Float16)0.f;
    }
    __syncthreads();

    // ---- folded tower GEMM: [16 x 128] x [128 x 32], bias in K row 100 ----
    {
        const v16h* wfr = (const v16h*)w2cf;
        v8f c0 = {}; v8f c1 = {};
        #pragma unroll
        for (int c = 0; c < 4; ++c) {
            v16h a = load_a_frag(sV + m16 * KP + c * 32, koff);
            v16h bt0 = wfr[(c * 2 + 0) * 32 + lane];
            v16h bt1 = wfr[(c * 2 + 1) * 32 + lane];
            c0 = __builtin_amdgcn_wmma_f32_16x16x32_f16(false, a, false, bt0, (short)0, c0, false, false);
            c1 = __builtin_amdgcn_wmma_f32_16x16x32_f16(false, a, false, bt1, (short)0, c1, false, false);
        }
        #pragma unroll
        for (int r = 0; r < 8; ++r) {
            int row = r + hi * 8;
            sH2[row * MPAD + m16]      = fmaxf(c0[r], 0.f);
            sH2[row * MPAD + 16 + m16] = fmaxf(c1[r], 0.f);
        }
    }
    __syncthreads();

    // ---- readout: out = relu(h2) @ w_out^T + b_out ----
    if (lane < NODES_PER_BLK) {
        float acc = b_out[0];
        #pragma unroll
        for (int f = 0; f < F1; ++f) acc += w_out[f] * sH2[lane * MPAD + f];
        out[nodeBase + lane] = acc;
    }
}

extern "C" void kernel_launch(void* const* d_in, const int* in_sizes, int n_in,
                              void* d_out, int out_size, void* d_ws, size_t ws_size,
                              hipStream_t stream) {
    const float* x        = (const float*)d_in[0];
    const int*   ei       = (const int*)d_in[1];
    const float* w_pre1   = (const float*)d_in[2];
    const float* b_pre1   = (const float*)d_in[3];
    const float* w_post1  = (const float*)d_in[4];
    const float* b_post1  = (const float*)d_in[5];
    const float* w_lin1   = (const float*)d_in[6];
    const float* b_lin1   = (const float*)d_in[7];
    const float* w_pre2   = (const float*)d_in[8];
    const float* b_pre2   = (const float*)d_in[9];
    const float* w_post2  = (const float*)d_in[10];
    const float* b_post2  = (const float*)d_in[11];
    const float* w_lin2   = (const float*)d_in[12];
    const float* b_lin2   = (const float*)d_in[13];
    const float* w_out    = (const float*)d_in[14];
    const float* b_out    = (const float*)d_in[15];

    const int N = in_sizes[0] / 3;
    const int E = in_sizes[1] / 2;
    const int* srcp = ei;
    const int* dstp = ei + E;

    char* ws = (char*)d_ws;
    float*    w1c   = (float*)(ws + WS_W1C);
    float*    b1c   = (float*)(ws + WS_B1C);
    _Float16* bpref = (_Float16*)(ws + WS_BPREF);
    _Float16* w2cf  = (_Float16*)(ws + WS_W2CF);
    _Float16* h1    = (_Float16*)(ws + WS_H1);

    pna_prep<<<1, 256, 0, stream>>>(w_post1, b_post1, w_lin1, b_lin1,
                                    w_pre2, b_pre2, w_post2, b_post2, w_lin2, b_lin2,
                                    w1c, b1c, bpref, w2cf);

    pna_layer1<<<(N + 255) / 256, 256, 0, stream>>>(x, srcp, w_pre1, b_pre1,
                                                    w1c, b1c, h1, N);

    pna_layer2<<<N / NODES_PER_BLK, 32, 0, stream>>>(srcp, dstp, h1, bpref, w2cf,
                                                     w_out, b_out, (float*)d_out);
}
